// QueryDecoderLayer_60361470378234
// MI455X (gfx1250) — matrix operand order repaired and checked by reading, not compiled, not run on hardware
//
#include <hip/hip_runtime.h>
#include <hip/hip_bf16.h>

typedef __hip_bfloat16 bf16;
typedef __attribute__((ext_vector_type(16))) __bf16 v16bf;
typedef __attribute__((ext_vector_type(8)))  float  v8f;

union Frag { v16bf v; uint4 q[2]; };

// ---------------------------------------------------------------------------
// Guarded load of 8 contiguous bf16 (edge tiles only): zero-fill out of range.
// ---------------------------------------------------------------------------
__device__ __forceinline__ uint4 ld8_guard(const unsigned short* rowp, int k, int Kd, bool ok) {
  unsigned int p0 = 0, p1 = 0, p2 = 0, p3 = 0;
  if (ok) {
    if ((k + 8) <= Kd) {
      return *(const uint4*)(rowp + k);
    }
    unsigned short t[8];
#pragma unroll
    for (int i = 0; i < 8; ++i) t[i] = ((k + i) < Kd) ? rowp[k + i] : (unsigned short)0;
    p0 = (unsigned)t[0] | ((unsigned)t[1] << 16);
    p1 = (unsigned)t[2] | ((unsigned)t[3] << 16);
    p2 = (unsigned)t[4] | ((unsigned)t[5] << 16);
    p3 = (unsigned)t[6] | ((unsigned)t[7] << 16);
  }
  return make_uint4(p0, p1, p2, p3);
}

// ---------------------------------------------------------------------------
// Generic batched NT GEMM:  C[m,n] = alpha * sum_k A[m,k] * B[n,k]
// A: bf16 [M x Kd] row stride lda (halves); B: bf16 [N x Kd] row stride ldb;
// C: f32 [M x N] row stride ldc.  Per-z bases: base = (z/zdiv)*s?h + (z%zdiv)*s?l.
// One wave computes a 16x32 tile of C (two 16x16 WMMA accumulators sharing the
// A fragment).  Interior tiles take a branch-free loop of global_load_b128 +
// v_wmma_f32_16x16x32_bf16; edge tiles take a guarded slow loop.
// ---------------------------------------------------------------------------
__global__ void __launch_bounds__(128)
k_gemm_nt(const bf16* __restrict__ A, const bf16* __restrict__ Bm, float* __restrict__ C,
          int M, int N, int Kd, int lda, int ldb, int ldc, float alpha,
          long long sAh, long long sAl, long long sBh, long long sBl,
          long long sCh, long long sCl, int zdiv)
{
  int z  = blockIdx.z;
  int zh = z / zdiv, zl = z - zh * zdiv;
  const unsigned short* Ap = (const unsigned short*)A + zh * sAh + zl * sAl;
  const unsigned short* Bp = (const unsigned short*)Bm + zh * sBh + zl * sBl;
  float* Cp = C + zh * sCh + zl * sCl;

  int lane = threadIdx.x & 31;
  int wave = threadIdx.x >> 5;
  int n0 = (blockIdx.x * 4 + wave) * 32;     // 2 N-tiles per wave
  if (n0 >= N) return;                       // wave-uniform exit
  int m0 = blockIdx.y * 16;

  int  arow  = m0 + (lane & 15);
  int  bcol0 = n0 + (lane & 15);
  int  bcol1 = n0 + 16 + (lane & 15);
  const unsigned short* arp  = Ap + (long long)arow * lda;
  const unsigned short* brp0 = Bp + (long long)bcol0 * ldb;
  const unsigned short* brp1 = Bp + (long long)bcol1 * ldb;
  int hiA = (lane >> 4) << 4;   // 0 or 16  (A: K groups of 16)
  int hiB = (lane >> 4) << 4;   // 0 or 16  (B: K groups of 16)

  v8f acc0 = {0.f, 0.f, 0.f, 0.f, 0.f, 0.f, 0.f, 0.f};
  v8f acc1 = {0.f, 0.f, 0.f, 0.f, 0.f, 0.f, 0.f, 0.f};

  bool interior = (m0 + 16 <= M) && (n0 + 32 <= N) && ((Kd & 31) == 0);
  if (interior) {
    // branch-free hot loop: pure vector loads + WMMA
    for (int k0 = 0; k0 < Kd; k0 += 32) {
      Frag fa, fb0, fb1;
      fa.q[0]  = *(const uint4*)(arp + k0 + hiA);
      fa.q[1]  = *(const uint4*)(arp + k0 + hiA + 8);
      fb0.q[0] = *(const uint4*)(brp0 + k0 + hiB);
      fb0.q[1] = *(const uint4*)(brp0 + k0 + hiB + 8);
      fb1.q[0] = *(const uint4*)(brp1 + k0 + hiB);
      fb1.q[1] = *(const uint4*)(brp1 + k0 + hiB + 8);
      __builtin_prefetch(arp + k0 + 256, 0, 0);   // global_prefetch_b8
      __builtin_prefetch(brp0 + k0 + 256, 0, 0);
      acc0 = __builtin_amdgcn_wmma_f32_16x16x32_bf16(false, fa.v, false, fb0.v,
                                                     (short)0, acc0, false, false);
      acc1 = __builtin_amdgcn_wmma_f32_16x16x32_bf16(false, fa.v, false, fb1.v,
                                                     (short)0, acc1, false, false);
    }
  } else {
    bool aOK  = arow < M;
    bool b0OK = bcol0 < N;
    bool b1OK = bcol1 < N;
    for (int k0 = 0; k0 < Kd; k0 += 32) {
      Frag fa, fb0, fb1;
      fa.q[0]  = ld8_guard(arp, k0 + hiA,      Kd, aOK);
      fa.q[1]  = ld8_guard(arp, k0 + hiA + 8,  Kd, aOK);
      fb0.q[0] = ld8_guard(brp0, k0 + hiB,     Kd, b0OK);
      fb0.q[1] = ld8_guard(brp0, k0 + hiB + 8, Kd, b0OK);
      fb1.q[0] = ld8_guard(brp1, k0 + hiB,     Kd, b1OK);
      fb1.q[1] = ld8_guard(brp1, k0 + hiB + 8, Kd, b1OK);
      acc0 = __builtin_amdgcn_wmma_f32_16x16x32_bf16(false, fa.v, false, fb0.v,
                                                     (short)0, acc0, false, false);
      acc1 = __builtin_amdgcn_wmma_f32_16x16x32_bf16(false, fa.v, false, fb1.v,
                                                     (short)0, acc1, false, false);
    }
  }

  // store (C layout: lane = N within tile, VGPR v = row m0+v (+8 for upper lanes))
  int rb = m0 + ((lane >> 4) << 3);
  int cn0 = n0 + (lane & 15);
  int cn1 = cn0 + 16;
  if (cn0 < N) {
#pragma unroll
    for (int v = 0; v < 8; ++v) {
      int r = rb + v;
      if (r < M) Cp[(long long)r * ldc + cn0] = alpha * acc0[v];
    }
  }
  if (cn1 < N) {
#pragma unroll
    for (int v = 0; v < 8; ++v) {
      int r = rb + v;
      if (r < M) Cp[(long long)r * ldc + cn1] = alpha * acc1[v];
    }
  }
}

// --------------------------- elementwise helpers ---------------------------
__global__ void k_cvt_bf16(const float* __restrict__ s, bf16* __restrict__ d, long long n) {
  long long st = (long long)gridDim.x * blockDim.x;
  for (long long i = (long long)blockIdx.x * blockDim.x + threadIdx.x; i < n; i += st)
    d[i] = __float2bfloat16(s[i]);
}

// padded convert: dst has row stride ldd >= cols; zero-fill cols..ldd-1
__global__ void k_cvt_pad(const float* __restrict__ s, bf16* __restrict__ d,
                          long long rows, int cols, int ldd) {
  long long n = rows * ldd;
  long long st = (long long)gridDim.x * blockDim.x;
  for (long long i = (long long)blockIdx.x * blockDim.x + threadIdx.x; i < n; i += st) {
    long long r = i / ldd; int j = (int)(i - r * ldd);
    d[i] = (j < cols) ? __float2bfloat16(s[r * cols + j]) : __float2bfloat16(0.f);
  }
}

__global__ void k_add_cvt(const float* __restrict__ a, const float* __restrict__ b,
                          bf16* __restrict__ d, long long n) {
  long long st = (long long)gridDim.x * blockDim.x;
  for (long long i = (long long)blockIdx.x * blockDim.x + threadIdx.x; i < n; i += st)
    d[i] = __float2bfloat16(a[i] + b[i]);
}

// act: 0 = none, 1 = exact gelu, 2 = sigmoid
__global__ void k_bias_act(float* __restrict__ Y, const float* __restrict__ bias,
                           bf16* __restrict__ out_bf, long long M, int N, int act) {
  long long n = M * N;
  long long st = (long long)gridDim.x * blockDim.x;
  for (long long i = (long long)blockIdx.x * blockDim.x + threadIdx.x; i < n; i += st) {
    float v = Y[i] + (bias ? bias[i % N] : 0.f);
    if (act == 1)      v = 0.5f * v * (1.f + erff(v * 0.70710678118654752f));
    else if (act == 2) v = 1.f / (1.f + expf(-v));
    Y[i] = v;
    if (out_bf) out_bf[i] = __float2bfloat16(v);
  }
}

__global__ void k_gate_qx(const float* __restrict__ q, const float* __restrict__ g,
                          const float* __restrict__ al, float* __restrict__ qx, long long n) {
  long long st = (long long)gridDim.x * blockDim.x;
  for (long long i = (long long)blockIdx.x * blockDim.x + threadIdx.x; i < n; i += st)
    qx[i] = q[i] + g[i] * al[i];
}

__global__ void k_concat(const float* __restrict__ q, const float* __restrict__ al,
                         bf16* __restrict__ out, long long n) {
  long long st = (long long)gridDim.x * blockDim.x;
  for (long long i = (long long)blockIdx.x * blockDim.x + threadIdx.x; i < n; i += st) {
    long long r = i >> 9; int c = (int)(i & 511);
    float v = (c < 256) ? q[r * 256 + c] : al[r * 256 + (c - 256)];
    out[i] = __float2bfloat16(v);
  }
}

// --------------------------- row-wise reductions ---------------------------
__global__ void __launch_bounds__(256)
k_rownorm_bf16(const float* __restrict__ X, bf16* __restrict__ out) { // cols == 256
  long long r = blockIdx.x; int tid = threadIdx.x;
  float v = X[r * 256 + tid];
  __shared__ float red[256];
  red[tid] = v * v; __syncthreads();
  for (int s = 128; s > 0; s >>= 1) { if (tid < s) red[tid] += red[tid + s]; __syncthreads(); }
  float inv = 1.f / fmaxf(sqrtf(red[0]), 1e-6f);
  out[r * 256 + tid] = __float2bfloat16(v * inv);
}

__global__ void __launch_bounds__(256)
k_add_ln(const float* __restrict__ x, const float* __restrict__ y,
         const float* __restrict__ g, const float* __restrict__ bb,
         float* __restrict__ out, bf16* __restrict__ out_bf) { // D == 256
  long long r = blockIdx.x; int tid = threadIdx.x;
  float v = x[r * 256 + tid] + y[r * 256 + tid];
  __shared__ float red[256];
  red[tid] = v; __syncthreads();
  for (int s = 128; s > 0; s >>= 1) { if (tid < s) red[tid] += red[tid + s]; __syncthreads(); }
  float mu = red[0] * (1.f / 256.f); __syncthreads();
  float d = v - mu;
  red[tid] = d * d; __syncthreads();
  for (int s = 128; s > 0; s >>= 1) { if (tid < s) red[tid] += red[tid + s]; __syncthreads(); }
  float var = red[0] * (1.f / 256.f);
  float o = d * rsqrtf(var + 1e-5f) * g[tid] + bb[tid];
  out[r * 256 + tid] = o;
  if (out_bf) out_bf[r * 256 + tid] = __float2bfloat16(o);
}

// softmax over `cols` of each row; optional bias (cross-attn: broadcast over heads)
__global__ void __launch_bounds__(256)
k_softmax_rows(float* __restrict__ S, const float* __restrict__ bias_sel,
               int cols, int Hn, int Qn, int CKn) {
  int r = blockIdx.x, tid = threadIdx.x;
  float* row = S + (long long)r * cols;
  const float* brow = nullptr;
  if (bias_sel) {
    int b = r / (Hn * Qn);
    int i = r % Qn;
    brow = bias_sel + ((long long)(b * Qn + i)) * CKn;
  }
  __shared__ float red[256];
  float mx = -3.4e38f;
  for (int j = tid; j < cols; j += 256) {
    float v = row[j] + (brow ? brow[j] : 0.f);
    row[j] = v;
    mx = fmaxf(mx, v);
  }
  red[tid] = mx; __syncthreads();
  for (int s = 128; s > 0; s >>= 1) { if (tid < s) red[tid] = fmaxf(red[tid], red[tid + s]); __syncthreads(); }
  mx = red[0]; __syncthreads();
  float sum = 0.f;
  for (int j = tid; j < cols; j += 256) { float e = expf(row[j] - mx); row[j] = e; sum += e; }
  red[tid] = sum; __syncthreads();
  for (int s = 128; s > 0; s >>= 1) { if (tid < s) red[tid] += red[tid + s]; __syncthreads(); }
  float inv = 1.f / red[0];
  for (int j = tid; j < cols; j += 256) row[j] *= inv;
}

// --------------------------- top-k machinery ------------------------------
// DESTRUCTIVE on sim: top-96 of 16384, softmax, weighted sum of memory rows.
__global__ void __launch_bounds__(256)
k_topk_align(float* __restrict__ sim, const float* __restrict__ memory,
             float* __restrict__ aligned) {
  int bq = blockIdx.x, tid = threadIdx.x;
  int b = bq / 200;
  float* row = sim + (long long)bq * 16384;
  __shared__ float red[256]; __shared__ int redi[256];
  __shared__ float vals[96]; __shared__ int idxs[96]; __shared__ float wts[96];
  for (int t = 0; t < 96; ++t) {
    float best = -3.4e38f; int bi = 0;
    for (int i = tid; i < 16384; i += 256) { float v = row[i]; if (v > best) { best = v; bi = i; } }
    red[tid] = best; redi[tid] = bi; __syncthreads();
    for (int s = 128; s > 0; s >>= 1) {
      if (tid < s && red[tid + s] > red[tid]) { red[tid] = red[tid + s]; redi[tid] = redi[tid + s]; }
      __syncthreads();
    }
    if (tid == 0) { vals[t] = red[0]; idxs[t] = redi[0]; row[redi[0]] = -3.4e38f; }
    __syncthreads();
  }
  if (tid == 0) {
    float mx = vals[0];
    for (int t = 1; t < 96; ++t) mx = fmaxf(mx, vals[t]);
    float s = 0.f;
    for (int t = 0; t < 96; ++t) { float e = expf(vals[t] - mx); wts[t] = e; s += e; }
    float inv = 1.f / s;
    for (int t = 0; t < 96; ++t) wts[t] *= inv;
  }
  __syncthreads();
  const float* mb = memory + (long long)b * 16384 * 256;
  float acc = 0.f;
  for (int t = 0; t < 96; ++t) acc += wts[t] * mb[(long long)idxs[t] * 256 + tid];
  aligned[(long long)bq * 256 + tid] = acc;
}

__global__ void __launch_bounds__(256)
k_colmax(const float* __restrict__ sim2, float* __restrict__ imp) { // grid (64, B)
  int b = blockIdx.y;
  int k = blockIdx.x * 256 + threadIdx.x;
  const float* base = sim2 + (long long)b * 200 * 16384 + k;
  float m = -3.4e38f;
  for (int q = 0; q < 200; ++q) m = fmaxf(m, base[(long long)q * 16384]);
  imp[b * 16384 + k] = m;
}

// DESTRUCTIVE on imp: top-1024 indices of 16384
__global__ void __launch_bounds__(256)
k_topk_idx(float* __restrict__ imp, int* __restrict__ kidx) { // grid B
  int b = blockIdx.x, tid = threadIdx.x;
  float* row = imp + b * 16384;
  __shared__ float red[256]; __shared__ int redi[256];
  for (int t = 0; t < 1024; ++t) {
    float best = -3.4e38f; int bi = 0;
    for (int i = tid; i < 16384; i += 256) { float v = row[i]; if (v > best) { best = v; bi = i; } }
    red[tid] = best; redi[tid] = bi; __syncthreads();
    for (int s = 128; s > 0; s >>= 1) {
      if (tid < s && red[tid + s] > red[tid]) { red[tid] = red[tid + s]; redi[tid] = redi[tid + s]; }
      __syncthreads();
    }
    if (tid == 0) { kidx[b * 1024 + t] = redi[0]; row[redi[0]] = -3.4e38f; }
    __syncthreads();
  }
}

// --------------------------- gathers / transpose ---------------------------
__global__ void k_gather_mem(const float* __restrict__ memory, const int* __restrict__ kidx,
                             bf16* __restrict__ msel) { // n = B*1024*256
  long long n = (long long)8 * 1024 * 256;
  long long st = (long long)gridDim.x * blockDim.x;
  for (long long i = (long long)blockIdx.x * blockDim.x + threadIdx.x; i < n; i += st) {
    int d = (int)(i & 255);
    long long rr = i >> 8;
    int j = (int)(rr & 1023);
    int b = (int)(rr >> 10);
    int src = kidx[b * 1024 + j];
    msel[i] = __float2bfloat16(memory[((long long)b * 16384 + src) * 256 + d]);
  }
}

__global__ void k_gather_bias(const float* __restrict__ cab, const int* __restrict__ kidx,
                              float* __restrict__ bsel) { // n = B*Q*1024
  long long n = (long long)8 * 200 * 1024;
  long long st = (long long)gridDim.x * blockDim.x;
  for (long long i = (long long)blockIdx.x * blockDim.x + threadIdx.x; i < n; i += st) {
    int j = (int)(i & 1023);
    long long rr = i >> 10;        // b*200 + q
    int b = (int)(rr / 200);
    bsel[i] = cab[rr * 16384 + kidx[b * 1024 + j]];
  }
}

// dst[b][h][d][j(0..Jp)] = j<J ? src[b*J + j][h*32 + d] : 0   (bf16, padded)
__global__ void k_transpose_v(const bf16* __restrict__ src, bf16* __restrict__ dst,
                              int J, int Jp) {
  long long n = (long long)8 * 8 * 32 * Jp;
  long long st = (long long)gridDim.x * blockDim.x;
  for (long long i = (long long)blockIdx.x * blockDim.x + threadIdx.x; i < n; i += st) {
    int j = (int)(i % Jp);
    long long r = i / Jp;
    int d = (int)(r & 31);
    int h = (int)((r >> 5) & 7);
    int b = (int)(r >> 8);
    dst[i] = (j < J) ? src[((long long)b * J + j) * 256 + h * 32 + d]
                     : __float2bfloat16(0.f);
  }
}

// ===========================================================================
extern "C" void kernel_launch(void* const* d_in, const int* in_sizes, int n_in,
                              void* d_out, int out_size, void* d_ws, size_t ws_size,
                              hipStream_t stream) {
  (void)in_sizes; (void)n_in; (void)out_size; (void)ws_size;
  const int Bn = 8, Qn = 200, Kn = 16384, Hn = 8, CKn = 1024;
  const int QP = 224;                         // Q padded to multiple of 32 (AV K-dim)
  const long long MBK = (long long)Bn * Kn;   // 131072
  const long long MBQ = (long long)Bn * Qn;   // 1600
  const float RS32 = 0.17677669529663687f;    // 1/sqrt(32)

  const float* q       = (const float*)d_in[0];
  const float* q_pos   = (const float*)d_in[1];
  const float* memory  = (const float*)d_in[2];
  const float* cab     = (const float*)d_in[3];

  // ---- workspace bump allocator ----
  char* wsp = (char*)d_ws; size_t off = 0;
  auto alloc = [&](size_t bytes) -> void* {
    void* p = wsp + off;
    off = (off + bytes + 255) & ~(size_t)255;
    return p;
  };
  bf16* wqA  = (bf16*)alloc(65536 * 2);
  bf16* wmA  = (bf16*)alloc(65536 * 2);
  bf16* wqC  = (bf16*)alloc(65536 * 2);
  bf16* wmC  = (bf16*)alloc(65536 * 2);
  bf16* saIn = (bf16*)alloc(196608 * 2);
  bf16* saOu = (bf16*)alloc(65536 * 2);
  bf16* caIn = (bf16*)alloc(196608 * 2);
  bf16* caOu = (bf16*)alloc(65536 * 2);
  bf16* fw1  = (bf16*)alloc(262144 * 2);
  bf16* fw2  = (bf16*)alloc(262144 * 2);
  bf16* gw1  = (bf16*)alloc(131072 * 2);
  bf16* gw2  = (bf16*)alloc(65536 * 2);
  bf16* mem_bf = (bf16*)alloc(MBK * 256 * 2);
  bf16* mn_bf  = (bf16*)alloc(MBK * 256 * 2);
  float* X     = (float*)alloc(MBK * 256 * 4);       // proj-out / sim / sim2 / reuse
  float* sbuf  = (float*)alloc(64LL * 200 * 1024 * 4);
  bf16* probs  = (bf16*)alloc(64LL * 200 * 1024 * 2);
  bf16* q_bf   = (bf16*)alloc(MBQ * 256 * 2);
  bf16* qn_bf  = (bf16*)alloc(MBQ * 256 * 2);
  float* alig  = (float*)alloc(MBQ * 256 * 4);
  bf16* cat_bf = (bf16*)alloc(MBQ * 512 * 2);
  float* tmpA  = (float*)alloc(MBQ * 1024 * 4);
  float* tmpB  = (float*)alloc(8192LL * 256 * 4);
  bf16* h_bf   = (bf16*)alloc(MBQ * 256 * 2);
  float* qx    = (float*)alloc(MBQ * 256 * 4);
  bf16* qx_bf  = (bf16*)alloc(MBQ * 256 * 2);
  bf16* qk_bf  = (bf16*)alloc(MBQ * 256 * 2);
  bf16* qp_bf  = (bf16*)alloc(MBQ * 256 * 2);
  bf16* kp_bf  = (bf16*)alloc(MBQ * 256 * 2);
  bf16* vp_bf  = (bf16*)alloc(MBQ * 256 * 2);
  bf16* vT_sa  = (bf16*)alloc(64LL * 32 * QP * 2);
  float* atto  = (float*)alloc(MBQ * 256 * 4);
  bf16* ao_bf  = (bf16*)alloc(MBQ * 256 * 2);
  bf16* msel   = (bf16*)alloc(8192LL * 256 * 2);
  bf16* kpc_bf = (bf16*)alloc(8192LL * 256 * 2);
  bf16* vpc_bf = (bf16*)alloc(8192LL * 256 * 2);
  bf16* vT_ca  = (bf16*)alloc(64LL * 32 * 1024 * 2);
  float* bsel  = (float*)alloc((long long)Bn * Qn * CKn * 4);
  float* imp   = (float*)alloc(Bn * Kn * 4);
  int*   kidx  = (int*)alloc(Bn * CKn * 4);
  bf16* f1_bf  = (bf16*)alloc(MBQ * 1024 * 2);

  auto cvt = [&](const void* s, bf16* d, long long n) {
    int blocks = (int)((n + 255) / 256); if (blocks > 65535) blocks = 65535;
    k_cvt_bf16<<<blocks, 256, 0, stream>>>((const float*)s, d, n);
  };
  auto gemm = [&](const bf16* A, const bf16* B, float* C, int M, int N, int Kd,
                  int lda, int ldb, int ldc, float alpha, int nz,
                  long long sAh, long long sAl, long long sBh, long long sBl,
                  long long sCh, long long sCl, int zdiv) {
    dim3 g((N + 127) / 128, (M + 15) / 16, nz);
    k_gemm_nt<<<g, 128, 0, stream>>>(A, B, C, M, N, Kd, lda, ldb, ldc, alpha,
                                     sAh, sAl, sBh, sBl, sCh, sCl, zdiv);
  };
  auto biasact = [&](float* Y, const void* b, bf16* obf, long long M, int N, int act) {
    long long n = M * N;
    int blocks = (int)((n + 255) / 256); if (blocks > 65535) blocks = 65535;
    k_bias_act<<<blocks, 256, 0, stream>>>(Y, (const float*)b, obf, M, N, act);
  };

  // ---- weight / input conversions ----
  cvt(d_in[4], wqA, 65536);   cvt(d_in[5], wmA, 65536);
  cvt(d_in[10], wqC, 65536);  cvt(d_in[11], wmC, 65536);
  cvt(d_in[12], saIn, 196608); cvt(d_in[14], saOu, 65536);
  cvt(d_in[16], caIn, 196608); cvt(d_in[18], caOu, 65536);
  cvt(d_in[20], fw1, 262144); cvt(d_in[22], fw2, 262144);
  cvt(d_in[6], gw1, 131072);  cvt(d_in[8], gw2, 65536);
  cvt(memory, mem_bf, MBK * 256);
  cvt(q, q_bf, MBQ * 256);

  // ---- align: mn = l2norm(memory @ Wm^T), qn = l2norm(q @ Wq^T) ----
  gemm(mem_bf, wmA, X, (int)MBK, 256, 256, 256, 256, 256, 1.f, 1, 0,0,0,0,0,0, 1);
  k_rownorm_bf16<<<(int)MBK, 256, 0, stream>>>(X, mn_bf);
  gemm(q_bf, wqA, tmpA, (int)MBQ, 256, 256, 256, 256, 256, 1.f, 1, 0,0,0,0,0,0, 1);
  k_rownorm_bf16<<<(int)MBQ, 256, 0, stream>>>(tmpA, qn_bf);

  // ---- sim [B,Q,K] + top-96 + softmax + weighted sum ----
  gemm(qn_bf, mn_bf, X, Qn, Kn, 256, 256, 256, Kn, 1.f, Bn,
       200LL*256, 0, (long long)Kn*256, 0, (long long)Qn*Kn, 0, 1);
  k_topk_align<<<(int)MBQ, 256, 0, stream>>>(X, memory, alig);

  // ---- gate MLP, qx = q + sigmoid(..)*aligned ----
  {
    long long n = MBQ * 512;
    k_concat<<<(int)((n + 255) / 256), 256, 0, stream>>>(q, alig, cat_bf, n);
  }
  gemm(cat_bf, gw1, tmpA, (int)MBQ, 256, 512, 512, 512, 256, 1.f, 1, 0,0,0,0,0,0, 1);
  biasact(tmpA, d_in[7], h_bf, MBQ, 256, 1);
  gemm(h_bf, gw2, tmpA, (int)MBQ, 256, 256, 256, 256, 256, 1.f, 1, 0,0,0,0,0,0, 1);
  biasact(tmpA, d_in[9], nullptr, MBQ, 256, 2);
  {
    long long n = MBQ * 256;
    k_gate_qx<<<(int)((n + 255) / 256), 256, 0, stream>>>(q, tmpA, alig, qx, n);
    k_cvt_bf16<<<(int)((n + 255) / 256), 256, 0, stream>>>(qx, qx_bf, n);
    k_add_cvt<<<(int)((n + 255) / 256), 256, 0, stream>>>(qx, q_pos, qk_bf, n);
  }

  // ---- self attention ----
  gemm(qk_bf, saIn,            tmpA, (int)MBQ, 256, 256, 256, 256, 256, 1.f, 1, 0,0,0,0,0,0, 1);
  biasact(tmpA, (const float*)d_in[13] + 0,   qp_bf, MBQ, 256, 0);
  gemm(qk_bf, saIn + 256*256,  tmpA, (int)MBQ, 256, 256, 256, 256, 256, 1.f, 1, 0,0,0,0,0,0, 1);
  biasact(tmpA, (const float*)d_in[13] + 256, kp_bf, MBQ, 256, 0);
  gemm(qx_bf, saIn + 512*256,  tmpA, (int)MBQ, 256, 256, 256, 256, 256, 1.f, 1, 0,0,0,0,0,0, 1);
  biasact(tmpA, (const float*)d_in[13] + 512, vp_bf, MBQ, 256, 0);
  {
    long long n = 64LL * 32 * QP;     // padded transpose: zero-fill j in [200,224)
    k_transpose_v<<<(int)((n + 255) / 256), 256, 0, stream>>>(vp_bf, vT_sa, Qn, QP);
  }
  gemm(qp_bf, kp_bf, sbuf, 200, 200, 32, 256, 256, 200, RS32, 64,
       51200, 32, 51200, 32, 320000, 40000, Hn);
  k_softmax_rows<<<64 * 200, 256, 0, stream>>>(sbuf, nullptr, 200, Hn, Qn, CKn);
  {
    long long n = 64LL * 200 * QP;    // padded probs (K-dim 200 -> 224, zeros)
    int blocks = (int)((n + 255) / 256); if (blocks > 65535) blocks = 65535;
    k_cvt_pad<<<blocks, 256, 0, stream>>>(sbuf, probs, 64LL * 200, 200, QP);
  }
  // AV: interior except the last M-tile (M = 200) (Kd = 224, multiple of 32; zeros contribute nothing)
  gemm(probs, vT_sa, atto, 200, 32, QP, QP, QP, 256, 1.f, 64,
       8LL*200*QP, 200LL*QP, 8LL*32*QP, 32LL*QP, 51200, 32, Hn);
  cvt(atto, ao_bf, MBQ * 256);
  gemm(ao_bf, saOu, tmpA, (int)MBQ, 256, 256, 256, 256, 256, 1.f, 1, 0,0,0,0,0,0, 1);
  biasact(tmpA, d_in[15], nullptr, MBQ, 256, 0);
  k_add_ln<<<(int)MBQ, 256, 0, stream>>>(qx, tmpA, (const float*)d_in[24],
                                         (const float*)d_in[25], qx, qx_bf);

  // ---- cross sparsify: sim2, importance, top-1024, gathers ----
  gemm(qx_bf, wqC, tmpA, (int)MBQ, 256, 256, 256, 256, 256, 1.f, 1, 0,0,0,0,0,0, 1);
  k_rownorm_bf16<<<(int)MBQ, 256, 0, stream>>>(tmpA, qn_bf);
  gemm(mem_bf, wmC, X, (int)MBK, 256, 256, 256, 256, 256, 1.f, 1, 0,0,0,0,0,0, 1);
  k_rownorm_bf16<<<(int)MBK, 256, 0, stream>>>(X, mn_bf);
  gemm(qn_bf, mn_bf, X, Qn, Kn, 256, 256, 256, Kn, 1.f, Bn,
       200LL*256, 0, (long long)Kn*256, 0, (long long)Qn*Kn, 0, 1);
  k_colmax<<<dim3(64, Bn), 256, 0, stream>>>(X, imp);
  k_topk_idx<<<Bn, 256, 0, stream>>>(imp, kidx);
  k_gather_mem<<<8192, 256, 0, stream>>>(memory, kidx, msel);
  k_gather_bias<<<6400, 256, 0, stream>>>(cab, kidx, bsel);

  // ---- cross attention ----
  {
    long long n = MBQ * 256;
    k_add_cvt<<<(int)((n + 255) / 256), 256, 0, stream>>>(qx, q_pos, qk_bf, n);
  }
  gemm(qk_bf, caIn,           tmpA, (int)MBQ, 256, 256, 256, 256, 256, 1.f, 1, 0,0,0,0,0,0, 1);
  biasact(tmpA, (const float*)d_in[17] + 0,   qp_bf, MBQ, 256, 0);
  gemm(msel, caIn + 256*256,  tmpB, 8192, 256, 256, 256, 256, 256, 1.f, 1, 0,0,0,0,0,0, 1);
  biasact(tmpB, (const float*)d_in[17] + 256, kpc_bf, 8192, 256, 0);
  gemm(msel, caIn + 512*256,  tmpB, 8192, 256, 256, 256, 256, 256, 1.f, 1, 0,0,0,0,0,0, 1);
  biasact(tmpB, (const float*)d_in[17] + 512, vpc_bf, 8192, 256, 0);
  {
    long long n = 64LL * 32 * 1024;
    k_transpose_v<<<(int)((n + 255) / 256), 256, 0, stream>>>(vpc_bf, vT_ca, 1024, 1024);
  }
  gemm(qp_bf, kpc_bf, sbuf, 200, 1024, 32, 256, 256, 1024, RS32, 64,
       51200, 32, 262144, 32, 1638400, 204800, Hn);
  k_softmax_rows<<<64 * 200, 256, 0, stream>>>(sbuf, bsel, 1024, Hn, Qn, CKn);
  cvt(sbuf, probs, 64LL * 200 * 1024);
  gemm(probs, vT_ca, atto, 200, 32, 1024, 1024, 1024, 256, 1.f, 64,
       1638400, 204800, 262144, 32768, 51200, 32, Hn);
  cvt(atto, ao_bf, MBQ * 256);
  gemm(ao_bf, caOu, tmpA, (int)MBQ, 256, 256, 256, 256, 256, 1.f, 1, 0,0,0,0,0,0, 1);
  biasact(tmpA, d_in[19], nullptr, MBQ, 256, 0);
  k_add_ln<<<(int)MBQ, 256, 0, stream>>>(qx, tmpA, (const float*)d_in[26],
                                         (const float*)d_in[27], qx, qx_bf);

  // ---- FFN + final LN ----
  gemm(qx_bf, fw1, tmpA, (int)MBQ, 1024, 256, 256, 256, 1024, 1.f, 1, 0,0,0,0,0,0, 1);
  biasact(tmpA, d_in[21], f1_bf, MBQ, 1024, 1);
  gemm(f1_bf, fw2, atto, (int)MBQ, 256, 1024, 1024, 1024, 256, 1.f, 1, 0,0,0,0,0,0, 1);
  biasact(atto, d_in[23], nullptr, MBQ, 256, 0);
  k_add_ln<<<(int)MBQ, 256, 0, stream>>>(qx, atto, (const float*)d_in[28],
                                         (const float*)d_in[29], (float*)d_out, nullptr);
}
